// PillarFeatureNet_59141699666577
// MI455X (gfx1250) — matrix operand order repaired and checked by reading, not compiled, not run on hardware
//
#include <hip/hip_runtime.h>
#include <hip/hip_bf16.h>

// ---------------- problem constants ----------------
#define NPTS   600000
#define BB     2
#define TT     5
#define GXc    512
#define GYc    512
#define CC     64
#define NSEG   (BB*TT*GYc*GXc)        // 2,621,440
#define CSTRIDE (TT*GYc*GXc)          // 1,310,720  (channel stride in output)
#define D_IN   9
#define EPSV   1e-3f

typedef __attribute__((ext_vector_type(16))) _Float16 v16h;
typedef __attribute__((ext_vector_type(8)))  float    v8f;

// ---------------------------------------------------------------------------
// Pass 1: per-point voxelization + scatter of counts / xyz sums + nvalid
// ---------------------------------------------------------------------------
__global__ __launch_bounds__(256) void k_points(const float* __restrict__ pts,
                                                float* __restrict__ counts,
                                                float* __restrict__ sums,
                                                int*   __restrict__ seg_arr,
                                                int*   __restrict__ pbase_arr,
                                                float* __restrict__ nvalid)
{
    __shared__ float s_cnt;
    if (threadIdx.x == 0) s_cnt = 0.0f;
    __syncthreads();

    int p = blockIdx.x * blockDim.x + threadIdx.x;
    bool valid = false;
    if (p < NPTS) {
        float bidf = pts[p*6+0];
        float x    = pts[p*6+1];
        float y    = pts[p*6+2];
        float z    = pts[p*6+3];
        float tf   = pts[p*6+5];
        float cxf = (x + 51.2f) * 5.0f;   // (x - pcr)/0.2
        float cyf = (y + 51.2f) * 5.0f;
        valid = (cxf >= 0.0f) && (cxf < (float)GXc) && (cyf >= 0.0f) && (cyf < (float)GYc);
        int cx = min(max((int)cxf, 0), GXc-1);
        int cy = min(max((int)cyf, 0), GYc-1);
        int bid = (int)bidf;
        int tid = (int)tf;
        int seg   = ((bid*TT + tid)*GYc + cy)*GXc + cx;
        int pbase = bid*(CC*CSTRIDE) + tid*(GYc*GXc) + cy*GXc + cx;
        seg_arr[p]   = valid ? seg   : -1;
        pbase_arr[p] = valid ? pbase : -1;
        if (valid) {
            atomicAdd(&counts[seg],   1.0f);
            atomicAdd(&sums[seg*3+0], x);
            atomicAdd(&sums[seg*3+1], y);
            atomicAdd(&sums[seg*3+2], z);
        }
    }
    if (valid) atomicAdd(&s_cnt, 1.0f);
    __syncthreads();
    if (threadIdx.x == 0) atomicAdd(nvalid, s_cnt);
}

// ---------------------------------------------------------------------------
// Build the 9 augmented features for point p (valid seg >= 0 assumed)
// ---------------------------------------------------------------------------
__device__ __forceinline__ void load_feats(const float* __restrict__ pts,
                                           const float* __restrict__ counts,
                                           const float* __restrict__ sums,
                                           int seg, int p, float f[9])
{
    float x  = pts[p*6+1];
    float y  = pts[p*6+2];
    float z  = pts[p*6+3];
    float it = pts[p*6+4];
    float cnt = fmaxf(counts[seg], 1.0f);
    float inv = 1.0f / cnt;
    float mx = sums[seg*3+0] * inv;
    float my = sums[seg*3+1] * inv;
    float mz = sums[seg*3+2] * inv;
    int cx = seg & (GXc-1);
    int cy = (seg >> 9) & (GYc-1);
    float centx = (float)cx * 0.2f + 0.1f - 51.2f;
    float centy = (float)cy * 0.2f + 0.1f - 51.2f;
    f[0] = x; f[1] = y; f[2] = z; f[3] = it;
    f[4] = x - mx; f[5] = y - my; f[6] = z - mz;
    f[7] = x - centx; f[8] = y - centy;
}

// ---------------------------------------------------------------------------
// WMMA core: build A (16x32 f16, contiguous k order, k = 16*(lane/16) + element) and run
// 4 WMMAs (one per 16-channel tile). Invalid points => zero A row.
// ---------------------------------------------------------------------------
__device__ __forceinline__ void gemm_tile(const float* __restrict__ pts,
                                          const float* __restrict__ counts,
                                          const float* __restrict__ sums,
                                          const float* __restrict__ sW,   // LDS 9x64
                                          int seg, int p, int hlf, int mrow,
                                          v8f acc[4])
{
    v16h A = {};
    if (seg >= 0) {
        float f[9];
        load_feats(pts, counts, sums, seg, p, f);
        if (hlf == 0) {
            #pragma unroll
            for (int e = 0; e < 9; ++e) A[e] = (_Float16)f[e];   // K=0..8
        } else {
            A[0] = (_Float16)f[8];                               // K=16 slot; B is zero there, no effect
        }
    }
    #pragma unroll
    for (int nt = 0; nt < 4; ++nt) {
        v16h Bv = {};
        if (hlf == 0) {   // lanes 0..15 hold K=0..15 of column N=mrow
            #pragma unroll
            for (int e = 0; e < D_IN; ++e)
                Bv[e] = (_Float16)sW[e*CC + nt*16 + mrow];
        }
        acc[nt] = __builtin_amdgcn_wmma_f32_16x16x32_f16(
            false, A, false, Bv, (short)0, acc[nt], false, false);
    }
}

// ---------------------------------------------------------------------------
// Pass 2: GEMM + per-channel sum / sumsq for BatchNorm (LDS-reduced)
// ---------------------------------------------------------------------------
__global__ __launch_bounds__(256) void k_gemm_stats(const float* __restrict__ pts,
                                                    const float* __restrict__ W,
                                                    const float* __restrict__ counts,
                                                    const float* __restrict__ sums,
                                                    const int*   __restrict__ seg_arr,
                                                    float* __restrict__ bn_sum,
                                                    float* __restrict__ bn_sq)
{
    __shared__ float sW[D_IN*CC];
    __shared__ float s_sum[CC];
    __shared__ float s_sq[CC];
    int t = threadIdx.x;
    for (int i = t; i < D_IN*CC; i += 256) sW[i] = W[i];
    if (t < CC) { s_sum[t] = 0.0f; s_sq[t] = 0.0f; }
    __syncthreads();

    int lane = t & 31;
    int hlf  = lane >> 4;
    int mrow = lane & 15;
    int wave = t >> 5;
    int tile = blockIdx.x * 8 + wave;
    bool active = (tile * 16 < NPTS);      // wave-uniform

    if (active) {
        int p = tile*16 + mrow;
        int seg = seg_arr[p];
        v8f acc[4] = {};
        gemm_tile(pts, counts, sums, sW, seg, p, hlf, mrow, acc);
        #pragma unroll
        for (int nt = 0; nt < 4; ++nt) {
            int c = nt*16 + mrow;
            float s = 0.0f, q = 0.0f;
            #pragma unroll
            for (int r = 0; r < 8; ++r) {
                float v = acc[nt][r];
                s += v;
                q += v * v;
            }
            atomicAdd(&s_sum[c], s);
            atomicAdd(&s_sq[c],  q);
        }
    }
    __syncthreads();
    if (t < CC) {
        atomicAdd(&bn_sum[t], s_sum[t]);
        atomicAdd(&bn_sq[t],  s_sq[t]);
    }
}

// ---------------------------------------------------------------------------
// Pass 3: fold BN into per-channel scale/bias
// ---------------------------------------------------------------------------
__global__ void k_bn(const float* __restrict__ bn_sum, const float* __restrict__ bn_sq,
                     const float* __restrict__ nvalid,
                     const float* __restrict__ gamma, const float* __restrict__ beta,
                     float* __restrict__ scale, float* __restrict__ bias)
{
    int c = threadIdx.x;
    if (c < CC) {
        float nv  = fmaxf(nvalid[0], 1.0f);
        float mu  = bn_sum[c] / nv;
        float var = fmaxf(bn_sq[c] / nv - mu*mu, 0.0f);
        float rs  = rsqrtf(var + EPSV);
        float a   = gamma[c] * rs;
        scale[c]  = a;
        bias[c]   = beta[c] - mu * a;
    }
}

// ---------------------------------------------------------------------------
// Pass 4: GEMM again + BN/ReLU + segment-max scatter into final layout
// relu output >= 0 so float-max == signed-int-bit max; canvas pre-zeroed.
// ---------------------------------------------------------------------------
__global__ __launch_bounds__(256) void k_gemm_scatter(const float* __restrict__ pts,
                                                      const float* __restrict__ W,
                                                      const float* __restrict__ counts,
                                                      const float* __restrict__ sums,
                                                      const int*   __restrict__ seg_arr,
                                                      const int*   __restrict__ pbase_arr,
                                                      const float* __restrict__ scale,
                                                      const float* __restrict__ bias,
                                                      float* __restrict__ out)
{
    __shared__ float sW[D_IN*CC];
    __shared__ int   s_pb[8][16];
    int t = threadIdx.x;
    for (int i = t; i < D_IN*CC; i += 256) sW[i] = W[i];

    int lane = t & 31;
    int hlf  = lane >> 4;
    int mrow = lane & 15;
    int wave = t >> 5;
    int tile = blockIdx.x * 8 + wave;
    bool active = (tile * 16 < NPTS);      // wave-uniform

    int p   = tile*16 + mrow;
    int seg = active ? seg_arr[p] : -1;
    if (active && hlf == 0) s_pb[wave][mrow] = pbase_arr[p];
    __syncthreads();

    if (active) {
        v8f acc[4] = {};
        gemm_tile(pts, counts, sums, sW, seg, p, hlf, mrow, acc);

        int* outi = (int*)out;
        #pragma unroll
        for (int nt = 0; nt < 4; ++nt) {
            int   c = nt*16 + mrow;
            float a = scale[c];
            float b = bias[c];
            #pragma unroll
            for (int r = 0; r < 8; ++r) {
                int m  = r + hlf*8;
                int pb = s_pb[wave][m];
                if (pb >= 0) {
                    float v = fmaxf(acc[nt][r] * a + b, 0.0f);
                    atomicMax(&outi[pb + c * CSTRIDE], __float_as_int(v));
                }
            }
        }
    }
}

// ---------------------------------------------------------------------------
extern "C" void kernel_launch(void* const* d_in, const int* in_sizes, int n_in,
                              void* d_out, int out_size, void* d_ws, size_t ws_size,
                              hipStream_t stream)
{
    const float* pts   = (const float*)d_in[0];   // (N, 6)
    const float* W     = (const float*)d_in[1];   // (9, 64)
    const float* gamma = (const float*)d_in[2];   // (64,)
    const float* beta  = (const float*)d_in[3];   // (64,)
    float* out = (float*)d_out;                   // (B, C, T, GY, GX)

    // workspace layout (floats)
    float* counts = (float*)d_ws;                 // NSEG
    float* sums   = counts + NSEG;                // NSEG*3
    float* bn_sum = sums + (size_t)NSEG*3;        // 64
    float* bn_sq  = bn_sum + CC;                  // 64
    float* nvalid = bn_sq + CC;                   // 1 (padded to 64)
    float* scale  = nvalid + CC;                  // 64
    float* bias   = scale + CC;                   // 64
    int*   seg_arr   = (int*)(bias + CC);         // N
    int*   pbase_arr = seg_arr + NPTS;            // N

    // zero canvas + stat accumulators (async memsets are graph-capturable)
    hipMemsetAsync(out, 0, (size_t)out_size * sizeof(float), stream);
    hipMemsetAsync(counts, 0, ((size_t)NSEG*4 + 3*CC) * sizeof(float), stream);

    k_points<<<(NPTS + 255)/256, 256, 0, stream>>>(pts, counts, sums,
                                                   seg_arr, pbase_arr, nvalid);

    int tiles  = (NPTS + 15) / 16;       // 37500
    int blocks = (tiles + 7) / 8;        // 8 waves (tiles) per 256-thread block

    k_gemm_stats<<<blocks, 256, 0, stream>>>(pts, W, counts, sums,
                                             seg_arr, bn_sum, bn_sq);

    k_bn<<<1, 64, 0, stream>>>(bn_sum, bn_sq, nvalid, gamma, beta, scale, bias);

    k_gemm_scatter<<<blocks, 256, 0, stream>>>(pts, W, counts, sums,
                                               seg_arr, pbase_arr,
                                               scale, bias, out);
}
